// Encoder_69793218560680
// MI455X (gfx1250) — matrix operand order repaired and checked by reading, not compiled, not run on hardware
//
#include <hip/hip_runtime.h>
#include <hip/hip_bf16.h>
#include <math.h>

#define D 512
#define H 8
#define DK 64
#define NL 6
#define DFF 2048
#define BATCH 2
#define SEQ 2048
#define MROWS (BATCH * SEQ)   // 4096

typedef __attribute__((ext_vector_type(16))) _Float16 v16h;
typedef __attribute__((ext_vector_type(8)))  _Float16 v8h;
typedef __attribute__((ext_vector_type(8)))  float    v8f;
typedef __attribute__((ext_vector_type(4)))  unsigned int u32x4;
typedef __attribute__((ext_vector_type(8)))  int i32x8;
typedef __attribute__((ext_vector_type(4)))  int i32x4;

// ---------------------------------------------------------------------------
// TDM: issue a 2D tensor_load_to_lds of a [rows=64] x [cols=32 halfs] tile of
// W[N,K] (f16 row-major) starting at (n0, kb) into LDS at lds_off.
// Descriptor packed per CDNA5 ISA 8.3/8.4.
// ---------------------------------------------------------------------------
__device__ __forceinline__ void tdm_load_w_tile(const _Float16* W, int N, int K,
                                                int n0, int kb,
                                                unsigned int lds_off) {
    unsigned long long ga =
        (unsigned long long)(uintptr_t)(W + (size_t)n0 * K + kb);

    u32x4 g0;
    g0.x = 1u;                                   // count=1, user mode, no gather
    g0.y = lds_off;                              // lds_addr
    g0.z = (unsigned int)ga;                     // global_addr[31:0]
    g0.w = (unsigned int)((ga >> 32) & 0x1FFFFFFu) | (2u << 30); // type=2

    unsigned int dim0 = (unsigned int)K;         // tensor_dim0 (elements)
    unsigned int dim1 = (unsigned int)N;         // tensor_dim1 (rows)
    unsigned int str0 = (unsigned int)K;         // tensor_dim0_stride (elements)

    i32x8 g1;
    g1[0] = 0x10000;                                             // data_size=1 (2B)
    g1[1] = (int)((dim0 & 0xFFFFu) << 16);                       // dim0[15:0] @ bits63:48
    g1[2] = (int)(((dim0 >> 16) & 0xFFFFu) | ((dim1 & 0xFFFFu) << 16));
    g1[3] = (int)(((dim1 >> 16) & 0xFFFFu) | (32u << 16));       // tile_dim0=32
    g1[4] = 64;                                                  // tile_dim1=64, tile_dim2=0
    g1[5] = (int)str0;                                           // dim0_stride lo32
    g1[6] = 0;                                                   // dim0_stride hi16, dim1_stride lo16
    g1[7] = 0;                                                   // dim1_stride hi32

    i32x4 z4 = {0, 0, 0, 0};
#if __clang_major__ >= 23
    i32x8 z8 = {0, 0, 0, 0, 0, 0, 0, 0};
    __builtin_amdgcn_tensor_load_to_lds(g0, g1, z4, z4, z8, 0);
#else
    __builtin_amdgcn_tensor_load_to_lds(g0, g1, z4, z4, 0);
#endif
}

// ---------------------------------------------------------------------------
// fp32 -> f16 conversion (weights, once per launch)
// ---------------------------------------------------------------------------
__global__ void cvt_f32_f16_kernel(const float* __restrict__ s,
                                   _Float16* __restrict__ d, size_t n) {
    size_t i = (size_t)blockIdx.x * 256 + threadIdx.x;
    if (i < n) d[i] = (_Float16)s[i];
}

// ---------------------------------------------------------------------------
// embedding + sinusoidal positions; writes fp32 x and f16 xh
// ---------------------------------------------------------------------------
__global__ void embed_kernel(const int* __restrict__ tok,
                             const float* __restrict__ emb,
                             float* __restrict__ x,
                             _Float16* __restrict__ xh) {
    size_t i = (size_t)blockIdx.x * 256 + threadIdx.x;   // over MROWS*D
    if (i >= (size_t)MROWS * D) return;
    int d  = (int)(i % D);
    size_t bl = i / D;
    int l  = (int)(bl % SEQ);
    int t  = tok[bl];
    int jj = (d / 2) * 2;
    float denom = powf(10000.0f, (float)jj / (float)D);
    float ang = (float)l / denom;
    float pos = (d & 1) ? cosf(ang) : sinf(ang);
    float v = emb[(size_t)t * D + d] + pos;
    x[i]  = v;
    xh[i] = (_Float16)v;
}

// ---------------------------------------------------------------------------
// GEMM: out[M,N] = act( A[M,K](f16) * W[N,K]^T (f16) + bias[N] )
// Block = 8 waves = 128(M) x 64(N); wave tile 16(M) x 64(N).
// Weight tile (64 rows x 32 halfs, 4 KB) staged to LDS by the Tensor Data
// Mover, double-buffered, synced with s_wait_tensorcnt + barriers.
// ACT: 0 = none, 1 = exact GELU
// ---------------------------------------------------------------------------
__device__ __forceinline__ float gelu_exact(float v) {
    return 0.5f * v * (1.0f + erff(v * 0.70710678118654752f));
}

template <int ACT>
__global__ __launch_bounds__(256) void gemm_wmma_kernel(
    const _Float16* __restrict__ A, const _Float16* __restrict__ W,
    const float* __restrict__ bias, float* __restrict__ outF,
    _Float16* __restrict__ outH, int M, int N, int K) {

    __shared__ _Float16 sB[2][64 * 32];          // double-buffered weight tile

    const int tid    = threadIdx.x;
    const int wave   = tid >> 5;
    const int lane   = tid & 31;
    const int lane15 = lane & 15;
    const int hb     = lane >> 4;                // half-wave index

    const int ntiles = N / 64;
    const int n0 = (blockIdx.x % ntiles) * 64;
    const int m0 = (blockIdx.x / ntiles) * 128 + wave * 16;

    const unsigned int lds0 = (unsigned int)(uintptr_t)(&sB[0][0]);
    const unsigned int lds1 = (unsigned int)(uintptr_t)(&sB[1][0]);

    v8f acc[4] = {};
    const _Float16* arow = A + (size_t)(m0 + lane15) * K;
    const int S = K / 32;

    if (wave == 0) tdm_load_w_tile(W, N, K, n0, 0, lds0);

    for (int s = 0; s < S; ++s) {
        const int cur = s & 1;
        if (wave == 0) {
            if (s + 1 < S) {
                tdm_load_w_tile(W, N, K, n0, (s + 1) * 32, cur ? lds0 : lds1);
                __builtin_amdgcn_s_wait_tensorcnt(1);
            } else {
                __builtin_amdgcn_s_wait_tensorcnt(0);
            }
        }
        __syncthreads();                         // weight tile visible to all waves

        const int kb = s * 32;
        __builtin_prefetch(arow + kb + 64, 0, 1);    // global_prefetch_b8 on A stream
        v16h a;
        const _Float16* ap = arow + kb + 8 * hb;
        ((v8h*)&a)[0] = *(const v8h*)(ap);
        ((v8h*)&a)[1] = *(const v8h*)(ap + 16);
#pragma unroll
        for (int j = 0; j < 4; ++j) {
            const _Float16* bp = &sB[cur][(16 * j + lane15) * 32 + 8 * hb];
            v16h bf;
            ((v8h*)&bf)[0] = *(const v8h*)bp;
            ((v8h*)&bf)[1] = *(const v8h*)(bp + 16);
            acc[j] = __builtin_amdgcn_wmma_f32_16x16x32_f16(
                false, a, false, bf, (short)0, acc[j], false, false);
        }
        __syncthreads();                         // reads done before next overwrite
    }

#pragma unroll
    for (int j = 0; j < 4; ++j) {
        int col = n0 + 16 * j + lane15;
        float bv = bias[col];
#pragma unroll
        for (int r = 0; r < 8; ++r) {
            int row = m0 + r + 8 * hb;
            float val = acc[j][r] + bv;
            if (ACT == 1) val = gelu_exact(val);
            size_t idx = (size_t)row * N + col;
            if (outF) outF[idx] = val;
            if (outH) outH[idx] = (_Float16)val;
        }
    }
}

// ---------------------------------------------------------------------------
// V transpose: v[bl, h*DK+d] -> vt[((b*H+h)*DK+d)*SEQ + l]
// ---------------------------------------------------------------------------
__global__ void vtrans_kernel(const _Float16* __restrict__ v,
                              _Float16* __restrict__ vt) {
    size_t i = (size_t)blockIdx.x * 256 + threadIdx.x;   // over MROWS*D
    if (i >= (size_t)MROWS * D) return;
    int d = (int)(i % D);
    size_t bl = i / D;
    int l = (int)(bl % SEQ);
    int b = (int)(bl / SEQ);
    int h  = d / DK;
    int dd = d % DK;
    vt[(((size_t)(b * H + h) * DK) + dd) * SEQ + l] = v[i];
}

// ---------------------------------------------------------------------------
// fused attention per (b, h, q-tile of 16):
//   WMMA scores -> scale/mask -> softmax (LDS) -> attn fp32 to d_out,
//   attn f16 to LDS -> WMMA context -> ctx f16
// block = 128 threads (4 waves); dyn LDS = 16*2048*4 + 16*2048*2 + 512 B
// ---------------------------------------------------------------------------
__global__ __launch_bounds__(128) void attn_kernel(
    const _Float16* __restrict__ qh, const _Float16* __restrict__ kh,
    const _Float16* __restrict__ vt, const int* __restrict__ tok,
    float* __restrict__ attn_out, _Float16* __restrict__ ctxh) {

    extern __shared__ char smem[];
    float*    sS   = (float*)smem;                                  // 16 x SEQ f32
    _Float16* sA   = (_Float16*)(smem + 16 * SEQ * 4);              // 16 x SEQ f16
    float*    sRed = (float*)(smem + 16 * SEQ * 4 + 16 * SEQ * 2);  // 16 x 8 f32

    const int tid    = threadIdx.x;
    const int wave   = tid >> 5;
    const int lane   = tid & 31;
    const int lane15 = lane & 15;
    const int hb     = lane >> 4;

    int bid = blockIdx.x;
    int qt = bid & (SEQ / 16 - 1);       // 0..127
    int h  = (bid >> 7) & (H - 1);
    int b  = bid >> 10;

    size_t qrow0 = (size_t)b * SEQ + qt * 16;

    // Q fragments: K = 64 -> two 16x16x32 steps, identical for all k-tiles
    v16h qa[2];
    {
        const _Float16* qp = qh + (qrow0 + lane15) * D + h * DK;
#pragma unroll
        for (int s = 0; s < 2; ++s) {
            ((v8h*)&qa[s])[0] = *(const v8h*)(qp + 32 * s + 8 * hb);
            ((v8h*)&qa[s])[1] = *(const v8h*)(qp + 32 * s + 8 * hb + 16);
        }
    }

    // -------- phase 1: score tiles (each wave strides over k-tiles) --------
    for (int kt = wave; kt < SEQ / 16; kt += 4) {
        v8f c = {};
        const _Float16* kp = kh + ((size_t)b * SEQ + kt * 16 + lane15) * D + h * DK;
#pragma unroll
        for (int s = 0; s < 2; ++s) {
            v16h bf;
            ((v8h*)&bf)[0] = *(const v8h*)(kp + 32 * s + 8 * hb);
            ((v8h*)&bf)[1] = *(const v8h*)(kp + 32 * s + 8 * hb + 16);
            c = __builtin_amdgcn_wmma_f32_16x16x32_f16(
                false, qa[s], false, bf, (short)0, c, false, false);
        }
        int colg = kt * 16 + lane15;
        bool pad = (tok[(size_t)b * SEQ + colg] == 0);
#pragma unroll
        for (int r = 0; r < 8; ++r) {
            float v = pad ? -INFINITY : c[r] * 0.125f;   // 1/sqrt(64)
            sS[(r + 8 * hb) * SEQ + colg] = v;
        }
    }
    __syncthreads();

    // -------- phase 2: softmax (16 row-teams of 8 threads) --------
    {
        int team = tid >> 3;
        int t8   = tid & 7;
        float m = -INFINITY;
        for (int c = t8; c < SEQ; c += 8) m = fmaxf(m, sS[team * SEQ + c]);
        sRed[team * 8 + t8] = m;
        __syncthreads();
        if (t8 == 0) {
            float mm = sRed[team * 8];
            for (int i = 1; i < 8; ++i) mm = fmaxf(mm, sRed[team * 8 + i]);
            sRed[team * 8] = mm;
        }
        __syncthreads();
        m = sRed[team * 8];
        __syncthreads();
        float s = 0.f;
        for (int c = t8; c < SEQ; c += 8) {
            float e = expf(sS[team * SEQ + c] - m);
            sS[team * SEQ + c] = e;
            s += e;
        }
        sRed[team * 8 + t8] = s;
        __syncthreads();
        if (t8 == 0) {
            float ss = 0.f;
            for (int i = 0; i < 8; ++i) ss += sRed[team * 8 + i];
            sRed[team * 8] = ss;
        }
        __syncthreads();
        float inv = 1.0f / sRed[team * 8];
        size_t obase = ((size_t)(b * H + h) * SEQ + (qt * 16 + team)) * SEQ;
        for (int c = t8; c < SEQ; c += 8) {
            float p = sS[team * SEQ + c] * inv;
            attn_out[obase + c] = p;
            sA[team * SEQ + c] = (_Float16)p;
        }
    }
    __syncthreads();

    // -------- phase 3: context (wave w owns d-tile w) --------
    {
        int d0 = wave * 16;
        v8f c = {};
        const _Float16* vbase =
            vt + (((size_t)(b * H + h) * DK) + d0 + lane15) * SEQ;
        for (int kb = 0; kb < SEQ; kb += 32) {
            v16h a;
            const _Float16* ap = sA + lane15 * SEQ + kb + 8 * hb;
            ((v8h*)&a)[0] = *(const v8h*)ap;
            ((v8h*)&a)[1] = *(const v8h*)(ap + 16);
            v16h bf;
            ((v8h*)&bf)[0] = *(const v8h*)(vbase + kb + 8 * hb);
            ((v8h*)&bf)[1] = *(const v8h*)(vbase + kb + 8 * hb + 16);
            c = __builtin_amdgcn_wmma_f32_16x16x32_f16(
                false, a, false, bf, (short)0, c, false, false);
        }
#pragma unroll
        for (int r = 0; r < 8; ++r) {
            size_t row = qrow0 + r + 8 * hb;
            ctxh[row * D + h * DK + d0 + lane15] = (_Float16)c[r];
        }
    }
}

// ---------------------------------------------------------------------------
// x = LayerNorm(a + x) * g + be ; writes fp32 x (in place) and f16 xh
// block = 256, one block per token row of D=512
// ---------------------------------------------------------------------------
__global__ __launch_bounds__(256) void add_ln_kernel(
    const float* __restrict__ a, float* __restrict__ x,
    const float* __restrict__ g, const float* __restrict__ be,
    _Float16* __restrict__ xh) {

    __shared__ float red[256];
    size_t row = blockIdx.x;
    int tid = threadIdx.x;
    size_t base = row * D;
    float v0 = a[base + tid] + x[base + tid];
    float v1 = a[base + tid + 256] + x[base + tid + 256];
    red[tid] = v0 + v1;
    __syncthreads();
    for (int o = 128; o > 0; o >>= 1) {
        if (tid < o) red[tid] += red[tid + o];
        __syncthreads();
    }
    float mu = red[0] * (1.0f / D);
    __syncthreads();
    float d0 = v0 - mu, d1 = v1 - mu;
    red[tid] = d0 * d0 + d1 * d1;
    __syncthreads();
    for (int o = 128; o > 0; o >>= 1) {
        if (tid < o) red[tid] += red[tid + o];
        __syncthreads();
    }
    float inv = rsqrtf(red[0] * (1.0f / D) + 1e-5f);
    float o0 = d0 * inv * g[tid] + be[tid];
    float o1 = d1 * inv * g[tid + 256] + be[tid + 256];
    x[base + tid] = o0;
    x[base + tid + 256] = o1;
    xh[base + tid] = (_Float16)o0;
    xh[base + tid + 256] = (_Float16)o1;
}

// ---------------------------------------------------------------------------
// host-side orchestration
// ---------------------------------------------------------------------------
extern "C" void kernel_launch(void* const* d_in, const int* in_sizes, int n_in,
                              void* d_out, int out_size, void* d_ws, size_t ws_size,
                              hipStream_t stream) {
    (void)in_sizes; (void)n_in; (void)out_size; (void)ws_size;

    const int*   tok   = (const int*)  d_in[0];
    const float* emb   = (const float*)d_in[1];
    const float* Wq    = (const float*)d_in[2];
    const float* bq    = (const float*)d_in[3];
    const float* Wk    = (const float*)d_in[4];
    const float* bk    = (const float*)d_in[5];
    const float* Wv    = (const float*)d_in[6];
    const float* bv    = (const float*)d_in[7];
    const float* Wo    = (const float*)d_in[8];
    const float* bo    = (const float*)d_in[9];
    const float* ln1g  = (const float*)d_in[10];
    const float* ln1b  = (const float*)d_in[11];
    const float* W1    = (const float*)d_in[12];
    const float* b1    = (const float*)d_in[13];
    const float* W2    = (const float*)d_in[14];
    const float* b2    = (const float*)d_in[15];
    const float* ln2g  = (const float*)d_in[16];
    const float* ln2b  = (const float*)d_in[17];

    // ---- workspace partition ----
    char* p = (char*)d_ws;
    auto take = [&](size_t bytes) -> char* {
        char* r = p;
        p += (bytes + 255) & ~(size_t)255;
        return r;
    };
    const size_t WQKVO_E = (size_t)NL * D * D;        // per-array halfs
    const size_t WFF_E   = (size_t)NL * DFF * D;
    _Float16* wq_h = (_Float16*)take(WQKVO_E * 2);
    _Float16* wk_h = (_Float16*)take(WQKVO_E * 2);
    _Float16* wv_h = (_Float16*)take(WQKVO_E * 2);
    _Float16* wo_h = (_Float16*)take(WQKVO_E * 2);
    _Float16* w1_h = (_Float16*)take(WFF_E * 2);
    _Float16* w2_h = (_Float16*)take(WFF_E * 2);
    float*    x_f  = (float*)   take((size_t)MROWS * D * 4);
    _Float16* x_h  = (_Float16*)take((size_t)MROWS * D * 2);
    _Float16* q_h  = (_Float16*)take((size_t)MROWS * D * 2);
    _Float16* k_h  = (_Float16*)take((size_t)MROWS * D * 2);
    _Float16* v_h  = (_Float16*)take((size_t)MROWS * D * 2);
    _Float16* vt_h = (_Float16*)take((size_t)MROWS * D * 2);
    _Float16* ctx_h= (_Float16*)take((size_t)MROWS * D * 2);
    float*    prj_f= (float*)   take((size_t)MROWS * D * 4);
    _Float16* hh_h = (_Float16*)take((size_t)MROWS * DFF * 2);
    float*    ffn_f= (float*)   take((size_t)MROWS * D * 4);

    // ---- weight conversions ----
    auto cvt = [&](const float* s, _Float16* d, size_t n) {
        cvt_f32_f16_kernel<<<(unsigned)((n + 255) / 256), 256, 0, stream>>>(s, d, n);
    };
    cvt(Wq, wq_h, WQKVO_E); cvt(Wk, wk_h, WQKVO_E);
    cvt(Wv, wv_h, WQKVO_E); cvt(Wo, wo_h, WQKVO_E);
    cvt(W1, w1_h, WFF_E);   cvt(W2, w2_h, WFF_E);

    // ---- embedding + positions ----
    {
        size_t n = (size_t)MROWS * D;
        embed_kernel<<<(unsigned)((n + 255) / 256), 256, 0, stream>>>(tok, emb, x_f, x_h);
    }

    const size_t X_ELEMS   = (size_t)MROWS * D;                 // 2,097,152
    const size_t ATT_ELEMS = (size_t)BATCH * H * SEQ * SEQ;     // 67,108,864
    const size_t ATT_SMEM  = 16 * SEQ * 4 + 16 * SEQ * 2 + 16 * 8 * 4;

    auto gemm_grid = [](int M, int N) -> unsigned {
        return (unsigned)((size_t)(M / 128) * (N / 64));
    };

    for (int i = 0; i < NL; ++i) {
        const _Float16* wqi = wq_h + (size_t)i * D * D;
        const _Float16* wki = wk_h + (size_t)i * D * D;
        const _Float16* wvi = wv_h + (size_t)i * D * D;
        const _Float16* woi = wo_h + (size_t)i * D * D;
        const _Float16* w1i = w1_h + (size_t)i * DFF * D;
        const _Float16* w2i = w2_h + (size_t)i * D * DFF;

        // Q, K, V projections (f16 outputs)
        gemm_wmma_kernel<0><<<gemm_grid(MROWS, D), 256, 0, stream>>>(
            x_h, wqi, bq + (size_t)i * D, nullptr, q_h, MROWS, D, D);
        gemm_wmma_kernel<0><<<gemm_grid(MROWS, D), 256, 0, stream>>>(
            x_h, wki, bk + (size_t)i * D, nullptr, k_h, MROWS, D, D);
        gemm_wmma_kernel<0><<<gemm_grid(MROWS, D), 256, 0, stream>>>(
            x_h, wvi, bv + (size_t)i * D, nullptr, v_h, MROWS, D, D);

        // V -> [b,h,d,l]
        {
            size_t n = (size_t)MROWS * D;
            vtrans_kernel<<<(unsigned)((n + 255) / 256), 256, 0, stream>>>(v_h, vt_h);
        }

        // fused attention: scores + softmax + context; attn weights to d_out
        float* attn_out = (float*)d_out + X_ELEMS + (size_t)i * ATT_ELEMS;
        attn_kernel<<<BATCH * H * (SEQ / 16), 128, ATT_SMEM, stream>>>(
            q_h, k_h, vt_h, tok, attn_out, ctx_h);

        // output projection (fp32)
        gemm_wmma_kernel<0><<<gemm_grid(MROWS, D), 256, 0, stream>>>(
            ctx_h, woi, bo + (size_t)i * D, prj_f, nullptr, MROWS, D, D);

        // x = LN(proj + x)
        add_ln_kernel<<<MROWS, 256, 0, stream>>>(
            prj_f, x_f, ln1g + (size_t)i * D, ln1b + (size_t)i * D, x_h);

        // FFN up with exact GELU (f16 output)
        gemm_wmma_kernel<1><<<gemm_grid(MROWS, DFF), 256, 0, stream>>>(
            x_h, w1i, b1 + (size_t)i * DFF, nullptr, hh_h, MROWS, DFF, D);

        // FFN down (fp32)
        gemm_wmma_kernel<0><<<gemm_grid(MROWS, D), 256, 0, stream>>>(
            hh_h, w2i, b2 + (size_t)i * D, ffn_f, nullptr, MROWS, D, DFF);

        // x = LN(ffn + x)
        add_ln_kernel<<<MROWS, 256, 0, stream>>>(
            ffn_f, x_f, ln2g + (size_t)i * D, ln2b + (size_t)i * D, x_h);
    }

    // final hidden states -> d_out[0 : X_ELEMS)
    hipMemcpyAsync(d_out, x_f, X_ELEMS * sizeof(float),
                   hipMemcpyDeviceToDevice, stream);
}
